// GNN_76639396430550
// MI455X (gfx1250) — matrix operand order repaired and checked by reading, not compiled, not run on hardware
//
#include <hip/hip_runtime.h>
#include <hip/hip_bf16.h>

// ---------------------------------------------------------------------------
// GCN (3x GCNConv + BN + ReLU) + segment_max + FC for MI455X (gfx1250).
//  - GEMMs: v_wmma_f32_16x16x32_f16; A operand stored f16 with padded rows so
//    the inner loop is pure clause'd b128 loads + WMMA; C rows padded to 32
//    floats so the epilogue stores are branchless.
//  - Edge aggregation: 128B-aligned float4 row gathers + L2-resident float
//    atomics (all node state << 192MB L2).
// ---------------------------------------------------------------------------

#define NNODES 200000            // divisible by 16 -> no M bounds checks
#define NEDGES 4000000
#define FIN    64
#define DIM    20
#define LDH    32                // padded f16 activation row stride
#define LDX    32                // padded f32 GEMM-output row stride (128B)
#define NGRAPH 512
#define NCLS   2
#define BNEPS  1e-5f

typedef __attribute__((ext_vector_type(16))) _Float16 v16h;
typedef __attribute__((ext_vector_type(8)))  _Float16 v8h;
typedef __attribute__((ext_vector_type(8)))  float    v8f;

// ---------------- degree / dinv ----------------
__global__ void deg_kernel(const int* __restrict__ ei, float* __restrict__ deg) {
    int e = blockIdx.x * blockDim.x + threadIdx.x;
    if (e >= NEDGES) return;
    atomicAdd(&deg[ei[e + NEDGES]], 1.0f);       // dst = edge_index[1][e]
}

__global__ void dinv_kernel(float* __restrict__ deg) {
    int i = blockIdx.x * blockDim.x + threadIdx.x;
    if (i >= NNODES) return;
    deg[i] = rsqrtf(1.0f + deg[i]);              // self loop => deg >= 1
}

// ---------------- f32 -> f16 convert (layer-0 input) ----------------
__global__ void cvt_f16(const float* __restrict__ src, _Float16* __restrict__ dst, int n) {
    int i = blockIdx.x * blockDim.x + threadIdx.x;
    if (i >= n) return;
    dst[i] = (_Float16)src[i];
}

// ---------------- pack W[K x 20] f32 into WMMA B-fragment layout -----------
// Bp[kb*1024 + ntile*512 + lane*16 + e] = f16(W[k*20 + n]), k = kb*32 + half*8 + (e<8 ? e : e+8),
// n = ntile*16 + (lane&15); zero outside K x 20.
__global__ void pack_b(const float* __restrict__ W, _Float16* __restrict__ Bp,
                       int K, int nkb) {
    int idx = blockIdx.x * blockDim.x + threadIdx.x;
    if (idx >= nkb * 1024) return;
    int kb    = idx >> 10;
    int r     = idx & 1023;
    int ntile = r >> 9;
    int lane  = (r >> 4) & 31;
    int e     = r & 15;
    int half  = lane >> 4;
    int n     = ntile * 16 + (lane & 15);
    int k     = kb * 32 + half * 8 + (e < 8 ? e : e + 8);
    float v   = (k < K && n < DIM) ? W[k * DIM + n] : 0.0f;
    Bp[idx] = (_Float16)v;
}

// ---------------- WMMA GEMM: C[N x LDX(pad)] = A16[N x LD] @ Bp ------------
// One wave per 16-row tile. A-fragment = two b128 loads per k-step
// (ISA 16-bit A layout: element e -> k = half*8 + (e<8 ? e : e+8)).
// C rows padded to LDX=32: both N-tiles stored unconditionally (branchless).
template <int LD, int NKB>
__global__ void gemm_wmma(const _Float16* __restrict__ A16,
                          const _Float16* __restrict__ Bp,
                          float* __restrict__ C) {
    int wave = blockIdx.x * (blockDim.x >> 5) + (threadIdx.x >> 5);
    int lane = threadIdx.x & 31;
    int m0 = wave * 16;
    if (m0 >= NNODES) return;                    // wave-uniform: EXEC all-1 at WMMA

    int half = lane >> 4;
    int row  = m0 + (lane & 15);
    const _Float16* arow = A16 + (size_t)row * LD + half * 8;

    v8f acc0 = {};
    v8f acc1 = {};
#pragma unroll
    for (int kb = 0; kb < NKB; ++kb) {
        v8h alo = *(const v8h*)(arow + kb * 32);         // k = kb + half*8 + 0..7
        v8h ahi = *(const v8h*)(arow + kb * 32 + 16);    // k = kb + half*8 + 16..23
        v16h a;
#pragma unroll
        for (int e = 0; e < 8; ++e) { a[e] = alo[e]; a[e + 8] = ahi[e]; }

        v16h b0 = *(const v16h*)(Bp + (size_t)kb * 1024 + lane * 16);
        v16h b1 = *(const v16h*)(Bp + (size_t)kb * 1024 + 512 + lane * 16);

        acc0 = __builtin_amdgcn_wmma_f32_16x16x32_f16(false, a, false, b0,
                                                      (short)0, acc0, false, false);
        acc1 = __builtin_amdgcn_wmma_f32_16x16x32_f16(false, a, false, b1,
                                                      (short)0, acc1, false, false);
    }

    // C/D layout: VGPR r -> row = half*8 + r, col = lane&15 (+16 for tile 1;
    // cols 20..31 land in padding, never read)
    int n = lane & 15;
#pragma unroll
    for (int r = 0; r < 8; ++r) {
        size_t base = (size_t)(m0 + half * 8 + r) * LDX;
        C[base + n]      = acc0[r];
        C[base + n + 16] = acc1[r];
    }
}

// ---------------- agg init: self-loop term + bias ----------------
__global__ void agg_init(const float* __restrict__ xl, const float* __restrict__ dinv,
                         const float* __restrict__ bias, float* __restrict__ agg) {
    int i = blockIdx.x * blockDim.x + threadIdx.x;
    if (i >= NNODES * DIM) return;
    int node = i / DIM;
    int d    = i - node * DIM;
    float di = dinv[node];
    agg[i] = xl[(size_t)node * LDX + d] * di * di + bias[d];
}

// ---------------- edge scatter-add: aligned float4 gather + atomics --------
__global__ void edge_agg(const int* __restrict__ ei, const float* __restrict__ xl,
                         const float* __restrict__ dinv, float* __restrict__ agg) {
    int e = blockIdx.x * blockDim.x + threadIdx.x;
    if (e >= NEDGES) return;
    int s = ei[e];
    int d = ei[e + NEDGES];
    float nrm = dinv[s] * dinv[d];
    const float4* xs = (const float4*)(xl + (size_t)s * LDX);   // 128B-aligned row
    float*        ad = agg + (size_t)d * DIM;
    float v[DIM];
#pragma unroll
    for (int q = 0; q < DIM / 4; ++q) {
        float4 t = xs[q];
        v[4 * q + 0] = t.x; v[4 * q + 1] = t.y;
        v[4 * q + 2] = t.z; v[4 * q + 3] = t.w;
    }
#pragma unroll
    for (int k = 0; k < DIM; ++k)
        atomicAdd(&ad[k], v[k] * nrm);
}

// ---------------- batchnorm: sum / sumsq per dim ----------------
__global__ void bn_reduce(const float* __restrict__ agg, float* __restrict__ stats) {
    __shared__ float sh[2 * DIM];
    for (int t = threadIdx.x; t < 2 * DIM; t += blockDim.x) sh[t] = 0.0f;
    __syncthreads();

    float s[DIM], q[DIM];
#pragma unroll
    for (int k = 0; k < DIM; ++k) { s[k] = 0.0f; q[k] = 0.0f; }

    int stride = gridDim.x * blockDim.x;
    for (int i = blockIdx.x * blockDim.x + threadIdx.x; i < NNODES; i += stride) {
        const float* r = agg + (size_t)i * DIM;
#pragma unroll
        for (int k = 0; k < DIM; ++k) { float v = r[k]; s[k] += v; q[k] += v * v; }
    }
#pragma unroll
    for (int k = 0; k < DIM; ++k) {
        atomicAdd(&sh[k], s[k]);
        atomicAdd(&sh[DIM + k], q[k]);
    }
    __syncthreads();
    for (int t = threadIdx.x; t < 2 * DIM; t += blockDim.x)
        atomicAdd(&stats[t], sh[t]);
}

// stats[0..20)=sum, [20..40)=sumsq -> stats[40..60)=scale, [60..80)=shift
__global__ void bn_finalize(float* __restrict__ stats, const float* __restrict__ g,
                            const float* __restrict__ beta) {
    int d = threadIdx.x;
    if (d >= DIM) return;
    const float invn = 1.0f / (float)NNODES;
    float mean = stats[d] * invn;
    float var  = stats[DIM + d] * invn - mean * mean;
    float sc   = g[d] * rsqrtf(var + BNEPS);
    stats[2 * DIM + d] = sc;
    stats[3 * DIM + d] = beta[d] - mean * sc;
}

// BN+ReLU -> padded f16 activations (next layer's WMMA A operand, ld = 32)
__global__ void bn_apply_relu_f16(const float* __restrict__ agg, const float* __restrict__ stats,
                                  _Float16* __restrict__ h16) {
    int i = blockIdx.x * blockDim.x + threadIdx.x;
    if (i >= NNODES * DIM) return;
    int node = i / DIM;
    int d    = i - node * DIM;
    float v = agg[i] * stats[2 * DIM + d] + stats[3 * DIM + d];
    v = v > 0.0f ? v : 0.0f;
    h16[(size_t)node * LDH + d] = (_Float16)v;
}

// BN+ReLU -> f32 node embeddings (final layer output)
__global__ void bn_apply_relu_f32(const float* __restrict__ agg, const float* __restrict__ stats,
                                  float* __restrict__ out) {
    int i = blockIdx.x * blockDim.x + threadIdx.x;
    if (i >= NNODES * DIM) return;
    int node = i / DIM;
    int d    = i - node * DIM;
    float v = agg[i] * stats[2 * DIM + d] + stats[3 * DIM + d];
    out[i] = v > 0.0f ? v : 0.0f;
}

// ---------------- segment_max (post-ReLU >= 0 -> int-punned max) ----------
__global__ void seg_max(const float* __restrict__ h, const int* __restrict__ batch,
                        float* __restrict__ ge) {
    int i = blockIdx.x * blockDim.x + threadIdx.x;
    if (i >= NNODES) return;
    int g = batch[i];
    int* gp = (int*)(ge + (size_t)g * DIM);
    const float* r = h + (size_t)i * DIM;
#pragma unroll
    for (int k = 0; k < DIM; ++k)
        atomicMax(&gp[k], __float_as_int(r[k]));
}

// ---------------- final FC [512,20] @ [20,2] + bias ----------------
__global__ void fc_kernel(const float* __restrict__ ge, const float* __restrict__ W,
                          const float* __restrict__ b, float* __restrict__ out) {
    int i = blockIdx.x * blockDim.x + threadIdx.x;
    if (i >= NGRAPH * NCLS) return;
    int g = i / NCLS;
    int c = i - g * NCLS;
    float acc = b[c];
#pragma unroll
    for (int k = 0; k < DIM; ++k)
        acc += ge[g * DIM + k] * W[k * NCLS + c];
    out[i] = acc;
}

// ---------------------------------------------------------------------------
extern "C" void kernel_launch(void* const* d_in, const int* in_sizes, int n_in,
                              void* d_out, int out_size, void* d_ws, size_t ws_size,
                              hipStream_t stream) {
    const float* x     = (const float*)d_in[0];
    const int*   ei    = (const int*)d_in[1];
    const int*   batch = (const int*)d_in[2];
    const float* W[3]  = {(const float*)d_in[3],  (const float*)d_in[7],  (const float*)d_in[11]};
    const float* bi[3] = {(const float*)d_in[4],  (const float*)d_in[8],  (const float*)d_in[12]};
    const float* g[3]  = {(const float*)d_in[5],  (const float*)d_in[9],  (const float*)d_in[13]};
    const float* be[3] = {(const float*)d_in[6],  (const float*)d_in[10], (const float*)d_in[14]};
    const float* fcW   = (const float*)d_in[15];
    const float* fcb   = (const float*)d_in[16];

    float* out      = (float*)d_out;
    float* node_emb = out;                                   // [N, DIM]
    float* ge       = out + (size_t)NNODES * DIM;            // [NG, DIM]
    float* logits   = ge  + (size_t)NGRAPH * DIM;            // [NG, NC]

    // workspace layout (f32 units; xl at 128B-aligned offset: N*4 % 128 == 0)
    float* ws    = (float*)d_ws;
    float* dinv  = ws;                                       // [N]
    float* xl    = dinv + NNODES;                            // [N, LDX] f32 padded
    float* agg   = xl + (size_t)NNODES * LDX;                // [N, DIM] f32
    float* stats = agg + (size_t)NNODES * DIM;               // [4*DIM]
    _Float16* x16  = (_Float16*)(stats + 128);               // [N, 64] f16 (layer 0 A)
    _Float16* h16  = x16;                                    // [N, 32] f16, aliases x16
                                                             // (x16 dead after gemm0)
    _Float16* Bp   = x16 + (size_t)NNODES * FIN;             // [2*1024] f16 packed B

    // degree / symmetric-normalization coefficients (edges fixed across layers)
    hipMemsetAsync(dinv, 0, (size_t)NNODES * sizeof(float), stream);
    deg_kernel<<<(NEDGES + 255) / 256, 256, 0, stream>>>(ei, dinv);
    dinv_kernel<<<(NNODES + 255) / 256, 256, 0, stream>>>(dinv);

    // layer-0 A operand: f16 copy of x
    cvt_f16<<<(NNODES * FIN + 255) / 256, 256, 0, stream>>>(x, x16, NNODES * FIN);

    const int gemmBlocks = ((NNODES / 16) + 3) / 4;          // 4 waves / 128-thr block
    const int elemBlocks = (NNODES * DIM + 255) / 256;

    for (int l = 0; l < 3; ++l) {
        int K   = (l == 0) ? FIN : DIM;
        int nkb = (l == 0) ? 2 : 1;
        pack_b<<<(nkb * 1024 + 255) / 256, 256, 0, stream>>>(W[l], Bp, K, nkb);
        if (l == 0) gemm_wmma<FIN, 2><<<gemmBlocks, 128, 0, stream>>>(x16, Bp, xl);
        else        gemm_wmma<LDH, 1><<<gemmBlocks, 128, 0, stream>>>(h16, Bp, xl);

        agg_init<<<elemBlocks, 256, 0, stream>>>(xl, dinv, bi[l], agg);
        edge_agg<<<(NEDGES + 255) / 256, 256, 0, stream>>>(ei, xl, dinv, agg);

        hipMemsetAsync(stats, 0, 2 * DIM * sizeof(float), stream);
        bn_reduce<<<512, 256, 0, stream>>>(agg, stats);
        bn_finalize<<<1, 32, 0, stream>>>(stats, g[l], be[l]);

        if (l == 0) {
            // zero padded f16 activation buffer once (pad cols 20..31 stay 0)
            hipMemsetAsync(h16, 0, (size_t)NNODES * LDH * sizeof(_Float16), stream);
        }
        if (l < 2) bn_apply_relu_f16<<<elemBlocks, 256, 0, stream>>>(agg, stats, h16);
        else       bn_apply_relu_f32<<<elemBlocks, 256, 0, stream>>>(agg, stats, node_emb);
    }

    hipMemsetAsync(ge, 0, (size_t)NGRAPH * DIM * sizeof(float), stream);
    seg_max<<<(NNODES + 255) / 256, 256, 0, stream>>>(node_emb, batch, ge);
    fc_kernel<<<(NGRAPH * NCLS + 255) / 256, 256, 0, stream>>>(ge, fcW, fcb, logits);
}
